// NNNet_31628139168544
// MI455X (gfx1250) — matrix operand order repaired and checked by reading, not compiled, not run on hardware
//
#include <hip/hip_runtime.h>
#include <hip/hip_bf16.h>
#include <cfloat>
#include <climits>

typedef __attribute__((ext_vector_type(16))) _Float16 v16h;
typedef __attribute__((ext_vector_type(8)))  _Float16 v8h;
typedef __attribute__((ext_vector_type(8)))  float    v8f;

#define NROWS_X   100000
#define NROWS_Q   2048
#define DIM       512
#define NCLASSES  1000
#define NTILES    (NROWS_X / 16)   // 6250 exact
#define KSTEPS    (DIM / 32)       // 16
#define COLGROUPS 512              // partial slots per query row (= gridDim.y)

#if __has_builtin(__builtin_amdgcn_sched_group_barrier)
#define SGB(mask, size, id) __builtin_amdgcn_sched_group_barrier(mask, size, id)
#else
#define SGB(mask, size, id)
#endif
#define SG_MFMA 0x008   // MFMA / WMMA
#define SG_VMRD 0x020   // VMEM read

__device__ __forceinline__ v16h pack16(v8h lo, v8h hi) {
    return __builtin_shufflevector(lo, hi,
        0, 1, 2, 3, 4, 5, 6, 7, 8, 9, 10, 11, 12, 13, 14, 15);
}

// ---------------------------------------------------------------------------
// Pass 1: L2-normalize rows (f32 in) -> f16 out. One wave per row.
// SWZ=0: row-major output (A operand / queries).
// SWZ=1: B-fragment order [tile][kstep][lane][16 halves].
// ---------------------------------------------------------------------------
template <bool SWZ>
__global__ __launch_bounds__(256) void l2norm_f16(const float* __restrict__ in,
                                                  _Float16* __restrict__ out,
                                                  int nrows) {
    const int w    = threadIdx.x >> 5;
    const int lane = threadIdx.x & 31;
    const int row  = blockIdx.x * 8 + w;
    if (row >= nrows) return;

    const float4* src = (const float4*)(in + (size_t)row * DIM) + (size_t)lane * 4;
    float4 a = src[0], b = src[1], c = src[2], d = src[3];

    float buf[16] = {a.x, a.y, a.z, a.w, b.x, b.y, b.z, b.w,
                     c.x, c.y, c.z, c.w, d.x, d.y, d.z, d.w};
    float s = 0.0f;
    #pragma unroll
    for (int i = 0; i < 16; ++i) s += buf[i] * buf[i];
    #pragma unroll
    for (int m = 1; m < 32; m <<= 1) s += __shfl_xor(s, m, 32);

    const float r = rsqrtf(fmaxf(s, 1e-24f));

    v8h lo, hi;
    #pragma unroll
    for (int i = 0; i < 8; ++i) {
        lo[i] = (_Float16)(buf[i] * r);
        hi[i] = (_Float16)(buf[i + 8] * r);
    }

    if (!SWZ) {
        v8h* dst = (v8h*)(out + (size_t)row * DIM + (size_t)lane * 16);
        dst[0] = lo;
        dst[1] = hi;
    } else {
        const int t  = row >> 4;
        const int nl = row & 15;
        const int ks = lane >> 1;
        const int hc = lane & 1;
        const size_t base =
            (((size_t)t * KSTEPS + ks) * 32 + (size_t)hc * 16 + nl) * 16;
        v8h* dst = (v8h*)(out + base);
        dst[0] = lo;
        dst[1] = hi;
    }
}

// ---------------------------------------------------------------------------
// Pass 2: WMMA GEMM (Q * Xn^T) + fused running argmax.
// One M-tile per wave (A in 128 VGPRs), two N-tiles in flight. The
// sched_group_barrier pattern pins a 2-K-step software pipeline:
// B loads run two steps ahead of the matrix pipe.
// ---------------------------------------------------------------------------
__global__ __launch_bounds__(256, 1) void knn_gemm_argmax(const _Float16* __restrict__ Qh,
                                                          const _Float16* __restrict__ Xs,
                                                          float* __restrict__ pval,
                                                          int* __restrict__ pidx) {
    const int w     = threadIdx.x >> 5;
    const int lane  = threadIdx.x & 31;
    const int half  = lane >> 4;
    const int nlane = lane & 15;
    const int mtile = blockIdx.x * 8 + w;

    // A tile (16x512 f16) -> 16 x v16h = 128 VGPRs, contiguous k order (k = 16*half + s).
    v16h Areg[KSTEPS];
    {
        const _Float16* qrow =
            Qh + ((size_t)mtile * 16 + nlane) * DIM + (size_t)half * 16;
        #pragma unroll
        for (int ks = 0; ks < KSTEPS; ++ks) {
            v8h alo = *(const v8h*)(qrow + ks * 32);
            v8h ahi = *(const v8h*)(qrow + ks * 32 + 8);
            Areg[ks] = pack16(alo, ahi);
        }
    }

    float bv[8];
    int   bi[8];
    #pragma unroll
    for (int r = 0; r < 8; ++r) { bv[r] = -FLT_MAX; bi[r] = INT_MAX; }

    const int slot = blockIdx.y;
    const size_t tile_halves = (size_t)KSTEPS * 32 * 16;   // 8192 halves / tile

    int t = slot;
    // ---- main loop: two N-tiles per iteration ----
    for (; t + COLGROUPS < NTILES; t += 2 * COLGROUPS) {
        const _Float16* b0 = Xs + (size_t)t * tile_halves + (size_t)lane * 16;
        const _Float16* b1 = b0 + (size_t)COLGROUPS * tile_halves;

        v8f a0 = {}, a1 = {};
        #pragma unroll
        for (int ks = 0; ks < KSTEPS; ++ks) {
            v8h l0 = *(const v8h*)(b0 + (size_t)ks * 512);
            v8h h0 = *(const v8h*)(b0 + (size_t)ks * 512 + 8);
            v8h l1 = *(const v8h*)(b1 + (size_t)ks * 512);
            v8h h1 = *(const v8h*)(b1 + (size_t)ks * 512 + 8);
            a0 = __builtin_amdgcn_wmma_f32_16x16x32_f16(
                false, Areg[ks], false, pack16(l0, h0), (short)0, a0, false, false);
            a1 = __builtin_amdgcn_wmma_f32_16x16x32_f16(
                false, Areg[ks], false, pack16(l1, h1), (short)0, a1, false, false);
        }
        // Scheduling pattern: loads stay 2 K-steps ahead of the WMMA pipe.
        // 64 VMEM reads / 32 WMMAs per body, exactly accounted:
        SGB(SG_VMRD, 8, 0);                     // K-steps 0,1 loads
        #pragma unroll
        for (int ks = 0; ks < KSTEPS - 2; ++ks) {
            SGB(SG_MFMA, 2, 0);                 // WMMAs for K-step ks
            SGB(SG_VMRD, 4, 0);                 // loads for K-step ks+2
        }
        SGB(SG_MFMA, 4, 0);                     // last two K-steps' WMMAs

        // lower-index tile first -> preserves first-occurrence tie rule
        const int n0 = t * 16 + nlane;
        const int n1 = (t + COLGROUPS) * 16 + nlane;
        #pragma unroll
        for (int r = 0; r < 8; ++r) {
            float v = a0[r];
            if (v > bv[r]) { bv[r] = v; bi[r] = n0; }
            v = a1[r];
            if (v > bv[r]) { bv[r] = v; bi[r] = n1; }
        }
    }
    // ---- tail: at most one remaining tile ----
    if (t < NTILES) {
        const _Float16* b0 = Xs + (size_t)t * tile_halves + (size_t)lane * 16;
        v8f a0 = {};
        #pragma unroll
        for (int ks = 0; ks < KSTEPS; ++ks) {
            v8h l0 = *(const v8h*)(b0 + (size_t)ks * 512);
            v8h h0 = *(const v8h*)(b0 + (size_t)ks * 512 + 8);
            a0 = __builtin_amdgcn_wmma_f32_16x16x32_f16(
                false, Areg[ks], false, pack16(l0, h0), (short)0, a0, false, false);
        }
        SGB(SG_VMRD, 4, 0);
        #pragma unroll
        for (int ks = 0; ks < KSTEPS - 2; ++ks) {
            SGB(SG_MFMA, 1, 0);
            SGB(SG_VMRD, 2, 0);
        }
        SGB(SG_MFMA, 2, 0);

        const int n0 = t * 16 + nlane;
        #pragma unroll
        for (int r = 0; r < 8; ++r) {
            const float v = a0[r];
            if (v > bv[r]) { bv[r] = v; bi[r] = n0; }
        }
    }

    // Reduce across the 16 lanes sharing a row (xor stays inside each half).
    #pragma unroll
    for (int r = 0; r < 8; ++r) {
        float v = bv[r];
        int   i = bi[r];
        #pragma unroll
        for (int m = 1; m < 16; m <<= 1) {
            const float ov = __shfl_xor(v, m, 32);
            const int   oi = __shfl_xor(i, m, 32);
            if (ov > v || (ov == v && oi < i)) { v = ov; i = oi; }
        }
        if (nlane == 0) {
            const int row = mtile * 16 + r + half * 8;
            pval[(size_t)row * COLGROUPS + slot] = v;
            pidx[(size_t)row * COLGROUPS + slot] = i;
        }
    }
}

// ---------------------------------------------------------------------------
// Pass 3: reduce 512 partials per query, gather label, emit one-hot row.
// ---------------------------------------------------------------------------
__global__ __launch_bounds__(256) void reduce_onehot(const float* __restrict__ pval,
                                                     const int* __restrict__ pidx,
                                                     const int* __restrict__ y,
                                                     float* __restrict__ out) {
    __shared__ float sv[256];
    __shared__ int   si[256];
    __shared__ int   label;

    const int row = blockIdx.x;
    const int tid = threadIdx.x;

    float bv = -FLT_MAX;
    int   bi = INT_MAX;
    for (int j = tid; j < COLGROUPS; j += 256) {
        const float v = pval[(size_t)row * COLGROUPS + j];
        const int   i = pidx[(size_t)row * COLGROUPS + j];
        if (v > bv || (v == bv && i < bi)) { bv = v; bi = i; }
    }
    sv[tid] = bv;
    si[tid] = bi;
    __syncthreads();
    for (int s = 128; s > 0; s >>= 1) {
        if (tid < s) {
            const float ov = sv[tid + s];
            const int   oi = si[tid + s];
            if (ov > sv[tid] || (ov == sv[tid] && oi < si[tid])) {
                sv[tid] = ov; si[tid] = oi;
            }
        }
        __syncthreads();
    }
    if (tid == 0) label = y[si[0]];
    __syncthreads();

    for (int c = tid; c < NCLASSES; c += 256)
        out[(size_t)row * NCLASSES + c] = (c == label) ? 1.0f : 0.0f;
}

// ---------------------------------------------------------------------------
extern "C" void kernel_launch(void* const* d_in, const int* in_sizes, int n_in,
                              void* d_out, int out_size, void* d_ws, size_t ws_size,
                              hipStream_t stream) {
    const float* x  = (const float*)d_in[0];   // [100000, 512] f32
    const int*   y  = (const int*)d_in[1];     // [100000] labels
    const float* xt = (const float*)d_in[2];   // [2048, 512] f32
    float* out = (float*)d_out;                // [2048, 1000] one-hot

    char* ws = (char*)d_ws;
    const size_t xh_bytes = (size_t)NROWS_X * DIM * sizeof(_Float16);
    const size_t qh_bytes = (size_t)NROWS_Q * DIM * sizeof(_Float16);
    const size_t pv_bytes = (size_t)NROWS_Q * COLGROUPS * sizeof(float);

    _Float16* Xs   = (_Float16*)(ws);                                 // swizzled B
    _Float16* Qh   = (_Float16*)(ws + xh_bytes);                      // row-major A
    float*    pval = (float*)(ws + xh_bytes + qh_bytes);
    int*      pidx = (int*)(ws + xh_bytes + qh_bytes + pv_bytes);

    l2norm_f16<true ><<<NROWS_X / 8, 256, 0, stream>>>(x,  Xs, NROWS_X);
    l2norm_f16<false><<<NROWS_Q / 8, 256, 0, stream>>>(xt, Qh, NROWS_Q);
    knn_gemm_argmax<<<dim3(NROWS_Q / 128, COLGROUPS), 256, 0, stream>>>(Qh, Xs, pval, pidx);
    reduce_onehot<<<NROWS_Q, 256, 0, stream>>>(pval, pidx, y, out);
}
